// GlmMLA_24756191494625
// MI455X (gfx1250) — matrix operand order repaired and checked by reading, not compiled, not run on hardware
//
#include <hip/hip_runtime.h>
#include <math.h>

// ---------------------------------------------------------------------------
// Types
// ---------------------------------------------------------------------------
typedef __bf16 bf16;
typedef __bf16 bf16x4  __attribute__((ext_vector_type(4)));
typedef __bf16 bf16x8  __attribute__((ext_vector_type(8)));
typedef __bf16 bf16x16 __attribute__((ext_vector_type(16)));
typedef float  f32x4   __attribute__((ext_vector_type(4)));
typedef float  f32x8   __attribute__((ext_vector_type(8)));

// D = A(16x32 bf16) * B(32x16 bf16) + C(16x16 f32)
__device__ __forceinline__ f32x8 wmma_bf16(bf16x16 a, bf16x16 b, f32x8 c) {
    return __builtin_amdgcn_wmma_f32_16x16x32_bf16(
        /*neg_a=*/false, a, /*neg_b=*/false, b,
        /*c_mod=*/(short)0, c, /*reuse_a=*/false, /*reuse_b=*/false);
}

// Async DMA: 16 bytes per lane, global -> LDS, tracked by ASYNCcnt.
// LDS offset = low 32 bits of the flat shared pointer (flat = {aperture, as3 off}).
__device__ __forceinline__ void async_copy_b128(void* lds_dst, const void* gsrc) {
    unsigned l = (unsigned)(size_t)lds_dst;
    asm volatile("global_load_async_to_lds_b128 %0, %1, off"
                 :: "v"(l), "v"(gsrc)
                 : "memory");
}
__device__ __forceinline__ void wait_async0() {
    asm volatile("s_wait_asynccnt 0" ::: "memory");
}

// A fragment (16x32, 16-bit): lane L holds row (L&15);
//   lanes 0-15: K in {kk*32+0..7, kk*32+16..23}; lanes 16-31: +8.
__device__ __forceinline__ bf16x16 load_frag_a(const bf16* p, long stride, int kk) {
    const int lane = threadIdx.x & 31;
    const bf16* q = p + (long)(lane & 15) * stride + kk * 32 + ((lane & 16) ? 8 : 0);
    bf16x8 lo = *(const bf16x8*)(q);
    bf16x8 hi = *(const bf16x8*)(q + 16);
    return __builtin_shufflevector(lo, hi, 0, 1, 2, 3, 4, 5, 6, 7,
                                           8, 9, 10, 11, 12, 13, 14, 15);
}

// B fragment (32x16, 16-bit): lane L holds column N=(L&15);
//   lanes 0-15: K in {0..7, 16..23}; lanes 16-31: +8.
__device__ __forceinline__ bf16x16 load_frag_b(const bf16* p, long stride) {
    const int lane = threadIdx.x & 31;
    const bf16* q = p + (long)((lane & 16) ? 8 : 0) * stride + (lane & 15);
    bf16x16 r;
#pragma unroll
    for (int e = 0; e < 8; ++e) {
        r[e]     = q[(long)e * stride];
        r[8 + e] = q[(long)(16 + e) * stride];
    }
    return r;
}

// ---------------------------------------------------------------------------
// fp32 -> bf16 bulk convert
// ---------------------------------------------------------------------------
__global__ __launch_bounds__(256) void f32_to_bf16_kernel(
    const float* __restrict__ in, bf16* __restrict__ out, long n) {
    long i = ((long)blockIdx.x * 256 + threadIdx.x) * 8;
    if (i + 8 > n) return;
    f32x4 a = *(const f32x4*)(in + i);
    f32x4 b = *(const f32x4*)(in + i + 4);
    bf16x8 r;
    r[0] = (bf16)a.x; r[1] = (bf16)a.y; r[2] = (bf16)a.z; r[3] = (bf16)a.w;
    r[4] = (bf16)b.x; r[5] = (bf16)b.y; r[6] = (bf16)b.z; r[7] = (bf16)b.w;
    *(bf16x8*)(out + i) = r;
}

// ---------------------------------------------------------------------------
// GEMM: C[M,N] = A_bf16[M,K] * W_f32[K, *], grouped weight column mapping:
//   weight_col(n) = (n>>7)*grp_wstride + (n&127)  (identity for grp_wstride==128)
// BM=128, BN=128, BK=64; 8 waves (2 M x 4 N), each wave 64x32 = 4x2 WMMA tiles.
// Double-buffered LDS; A tile staged via async-to-LDS DMA, B tile converted
// f32->bf16 through VGPRs.
// ---------------------------------------------------------------------------
#define G_BM 128
#define G_BN 128
#define G_BK 64
#define G_SA 88    // LDS row stride (bf16) for A tile: 176B, 16B aligned
#define G_SB 136   // LDS row stride (bf16) for B tile: 272B, 16B aligned

template <typename OUT>
__global__ __launch_bounds__(256) void gemm_bf16_kernel(
    const bf16* __restrict__ A, long lda,
    const float* __restrict__ W, long ldw, int grp_wstride,
    OUT* __restrict__ C, long ldc, int K) {
    __shared__ bf16 sA[2][G_BM * G_SA];
    __shared__ bf16 sB[2][G_BK * G_SB];

    const int t = threadIdx.x;
    const int lane = t & 31;
    const int w = t >> 5;
    const long m0 = (long)blockIdx.y * G_BM;
    const long n0 = (long)blockIdx.x * G_BN;
    const int wm = (w >> 2) * 64;  // wave M offset inside tile
    const int wn = (w & 3) * 32;   // wave N offset inside tile

    f32x8 acc[4][2];
#pragma unroll
    for (int i = 0; i < 4; ++i)
#pragma unroll
        for (int j = 0; j < 2; ++j) acc[i][j] = (f32x8)0.0f;

    f32x4 breg[8];

    // A tile: 128x64 bf16 = 1024 x 16B chunks; 4 async DMAs per thread.
    auto stage_a = [&](int buf, int k0) {
#pragma unroll
        for (int r2 = 0; r2 < 4; ++r2) {
            int c = t + r2 * 256;          // chunk id 0..1023
            int row = c >> 3;              // 0..127
            int col = (c & 7) * 8;         // 0..56 (bf16 elems)
            async_copy_b128(&sA[buf][row * G_SA + col],
                            A + (m0 + row) * lda + k0 + col);
        }
    };
    // B tile: 64x128 f32, 4 elems/thread/iter, 8 iters -> registers
    auto load_b = [&](int k0) {
#pragma unroll
        for (int it = 0; it < 8; ++it) {
            int idx = t + it * 256;        // 0..2047
            int kr  = idx >> 5;            // 0..63
            int nc  = (idx & 31) * 4;      // 0..124
            long n  = n0 + nc;
            long wcol = (n >> 7) * (long)grp_wstride + (n & 127);
            breg[it] = *(const f32x4*)(W + (long)(k0 + kr) * ldw + wcol);
        }
    };
    auto store_b = [&](int buf) {
#pragma unroll
        for (int it = 0; it < 8; ++it) {
            int idx = t + it * 256;
            int kr  = idx >> 5;
            int nc  = (idx & 31) * 4;
            bf16x4 bv;
            bv.x = (bf16)breg[it].x; bv.y = (bf16)breg[it].y;
            bv.z = (bf16)breg[it].z; bv.w = (bf16)breg[it].w;
            *(bf16x4*)(&sB[buf][kr * G_SB + nc]) = bv;
        }
    };

    const int NK = K / G_BK;
    // prologue: fill buffer 0
    stage_a(0, 0);
    load_b(0);
    store_b(0);
    wait_async0();
    __syncthreads();

    for (int kt = 0; kt < NK; ++kt) {
        const int cur = kt & 1;
        if (kt + 1 < NK) {  // kick off next tile before computing this one
            stage_a(cur ^ 1, (kt + 1) * G_BK);
            load_b((kt + 1) * G_BK);
        }
#pragma unroll
        for (int kk = 0; kk < 2; ++kk) {
            bf16x16 af[4], bfr[2];
#pragma unroll
            for (int i = 0; i < 4; ++i)
                af[i] = load_frag_a(&sA[cur][(wm + 16 * i) * G_SA], G_SA, kk);
#pragma unroll
            for (int j = 0; j < 2; ++j)
                bfr[j] = load_frag_b(&sB[cur][(kk * 32) * G_SB + (wn + 16 * j)], G_SB);
#pragma unroll
            for (int i = 0; i < 4; ++i)
#pragma unroll
                for (int j = 0; j < 2; ++j)
                    acc[i][j] = wmma_bf16(af[i], bfr[j], acc[i][j]);
        }
        if (kt + 1 < NK) {
            store_b(cur ^ 1);
            wait_async0();
            __syncthreads();
        }
    }

    // C layout: lanes 0-15 hold N=lane rows 0-7; lanes 16-31 hold N=lane-16 rows 8-15
    const int roff = (lane & 16) ? 8 : 0;
    const int ncol = lane & 15;
#pragma unroll
    for (int i = 0; i < 4; ++i)
#pragma unroll
        for (int j = 0; j < 2; ++j) {
            long r = m0 + wm + 16 * i + roff;
            long c = n0 + wn + 16 * j + ncol;
#pragma unroll
            for (int e = 0; e < 8; ++e)
                C[(r + e) * ldc + c] = (OUT)acc[i][j][e];
        }
}

// ---------------------------------------------------------------------------
// Flash-style causal attention for one (batch, head, 128-row q tile).
// Q: bf16 [B*S, 4096] (col = h*128 + d), KV: bf16 [B*S, 8192]
//   (K at col h*128+d, V at col 4096 + h*128 + d). O: bf16 [B*S, 4096].
// 8 waves; wave w owns q rows [q0+16w, q0+16w+16). Key blocks of 32.
// V tile is staged by async DMA; K tile is transposed through VGPRs.
// ---------------------------------------------------------------------------
#define AT_SKT 40    // sKT [128 dims][32 keys] row stride
#define AT_SV  136   // sV  [32 keys][128 dims] row stride
#define AT_SP  40    // sP per-wave [16 q][32 keys] row stride

__global__ __launch_bounds__(256) void mla_attention_kernel(
    const bf16* __restrict__ Q, const bf16* __restrict__ KV,
    bf16* __restrict__ O, float scale, int S) {
    __shared__ bf16 sKT[128 * AT_SKT];
    __shared__ bf16 sV[32 * AT_SV];
    __shared__ bf16 sP[8 * 16 * AT_SP];

    const int t = threadIdx.x, lane = t & 31, w = t >> 5;
    const int b = blockIdx.z, h = blockIdx.y;
    const int q0 = blockIdx.x * 128;
    const int qr = q0 + w * 16;                // wave's first q row (in-seq)
    const long qrow_g = (long)b * S + qr;      // global row index

    // Q fragments: 4 fragments of K=32 covering head dims 0..127
    bf16x16 qf[4];
#pragma unroll
    for (int f = 0; f < 4; ++f)
        qf[f] = load_frag_a(Q + qrow_g * 4096 + h * 128, 4096, f);

    float m[8], l[8];
    f32x8 o[8];
#pragma unroll
    for (int e = 0; e < 8; ++e) { m[e] = -3e38f; l[e] = 0.0f; }
#pragma unroll
    for (int d = 0; d < 8; ++d) o[d] = (f32x8)0.0f;

    const int key_row = t >> 3;          // 0..31 (staging)
    const int dimb = (t & 7) * 16;       // 0..112

    const int jend = q0 + 128;
    for (int j = 0; j < jend; j += 32) {
        __syncthreads();
        {   // cooperative stage of V (async DMA) and K (transposed) tiles
            const long krow = (long)b * S + j + key_row;
            const bf16* vp = KV + krow * 8192 + 4096 + h * 128 + dimb;
            async_copy_b128(sV + key_row * AT_SV + dimb, vp);
            async_copy_b128(sV + key_row * AT_SV + dimb + 8, vp + 8);

            const bf16* kp = KV + krow * 8192 + h * 128 + dimb;
            bf16x8 k0v = *(const bf16x8*)(kp);
            bf16x8 k1v = *(const bf16x8*)(kp + 8);
#pragma unroll
            for (int e = 0; e < 8; ++e) sKT[(dimb + e) * AT_SKT + key_row] = k0v[e];
#pragma unroll
            for (int e = 0; e < 8; ++e) sKT[(dimb + 8 + e) * AT_SKT + key_row] = k1v[e];
        }
        wait_async0();
        __syncthreads();
        if (j > qr + 15) continue;  // this wave's rows precede the block

        // --- scores: S(16x32) = Q(16x128) * K^T(128x32), two 16x16 C tiles ---
        f32x8 s0 = (f32x8)0.0f, s1 = (f32x8)0.0f;
#pragma unroll
        for (int f = 0; f < 4; ++f) {
            bf16x16 bk0 = load_frag_b(sKT + (f * 32) * AT_SKT, AT_SKT);
            bf16x16 bk1 = load_frag_b(sKT + (f * 32) * AT_SKT + 16, AT_SKT);
            s0 = wmma_bf16(qf[f], bk0, s0);
            s1 = wmma_bf16(qf[f], bk1, s1);
        }

        // --- scale + causal mask ---
        const int rbase = qr + ((lane & 16) ? 8 : 0);
        const int key0 = j + (lane & 15);
        float alpha[8];
#pragma unroll
        for (int e = 0; e < 8; ++e) {
            const int r = rbase + e;
            s0[e] = (key0      <= r) ? s0[e] * scale : -3e38f;
            s1[e] = (key0 + 16 <= r) ? s1[e] * scale : -3e38f;
        }
        // --- online softmax (row reductions across 16-lane halves) ---
#pragma unroll
        for (int e = 0; e < 8; ++e) {
            float mx = fmaxf(s0[e], s1[e]);
            mx = fmaxf(mx, __shfl_xor(mx, 8, 32));
            mx = fmaxf(mx, __shfl_xor(mx, 4, 32));
            mx = fmaxf(mx, __shfl_xor(mx, 2, 32));
            mx = fmaxf(mx, __shfl_xor(mx, 1, 32));
            const float mn = fmaxf(m[e], mx);
            const float a0 = __expf(m[e] - mn);
            const float p0 = __expf(s0[e] - mn);
            const float p1 = __expf(s1[e] - mn);
            float rs = p0 + p1;
            rs += __shfl_xor(rs, 8, 32);
            rs += __shfl_xor(rs, 4, 32);
            rs += __shfl_xor(rs, 2, 32);
            rs += __shfl_xor(rs, 1, 32);
            l[e] = l[e] * a0 + rs;
            m[e] = mn;
            alpha[e] = a0;
            s0[e] = p0;
            s1[e] = p1;
        }
        // rescale accumulated output
#pragma unroll
        for (int d = 0; d < 8; ++d)
#pragma unroll
            for (int e = 0; e < 8; ++e) o[d][e] *= alpha[e];

        // --- P (C layout, f32) -> A-fragment (bf16) via per-wave LDS region ---
        bf16* sp = sP + w * 16 * AT_SP;
        const int prow = (lane & 16) ? 8 : 0;
#pragma unroll
        for (int e = 0; e < 8; ++e) {
            sp[(prow + e) * AT_SP + (lane & 15)]      = (bf16)s0[e];
            sp[(prow + e) * AT_SP + 16 + (lane & 15)] = (bf16)s1[e];
        }
        // DS ops are in-order within a wave: read-back sees the stores.
        bf16x16 pf = load_frag_a(sp, AT_SP, 0);
        // --- O += P(16x32) * V(32x128): 8 WMMA over dim tiles ---
#pragma unroll
        for (int d = 0; d < 8; ++d) {
            bf16x16 bv = load_frag_b(sV + d * 16, AT_SV);
            o[d] = wmma_bf16(pf, bv, o[d]);
        }
    }

    // epilogue: O_row = acc_row / l_row
    const int roff = (lane & 16) ? 8 : 0;
    const int ncol = lane & 15;
#pragma unroll
    for (int e = 0; e < 8; ++e) {
        const float inv = 1.0f / l[e];
        const long r = qrow_g + roff + e;
#pragma unroll
        for (int d = 0; d < 8; ++d)
            O[r * 4096 + h * 128 + d * 16 + ncol] = (bf16)(o[d][e] * inv);
    }
}

// ---------------------------------------------------------------------------
// Launch. Workspace layout (assumes ws_size >= 160 MiB):
//   [0,32M)     hs_bf16   [4096, 4096]
//   [32M,64M)   q_bf16    [4096, 4096]  (nope part only, grouped cols of wq)
//   [64M,128M)  kv_bf16   [4096, 8192]  (k | v, rope cols skipped)
//   [128M,160M) attn_bf16 [4096, 4096]
// ---------------------------------------------------------------------------
extern "C" void kernel_launch(void* const* d_in, const int* in_sizes, int n_in,
                              void* d_out, int out_size, void* d_ws, size_t ws_size,
                              hipStream_t stream) {
    (void)in_sizes; (void)n_in; (void)out_size; (void)ws_size;
    const float* hs  = (const float*)d_in[0];   // [4, 1024, 4096]
    const float* wq  = (const float*)d_in[1];   // [4096, 6144]
    const float* wkv = (const float*)d_in[2];   // [4096, 8256]
    const float* wo  = (const float*)d_in[3];   // [4096, 4096]
    float* out = (float*)d_out;                 // [4, 1024, 4096]

    char* ws = (char*)d_ws;
    bf16* hsb  = (bf16*)(ws);
    bf16* qb   = (bf16*)(ws + ((size_t)32 << 20));
    bf16* kvb  = (bf16*)(ws + ((size_t)64 << 20));
    bf16* attn = (bf16*)(ws + ((size_t)128 << 20));

    const int B = 4, S = 1024, HID = 4096;
    const int M = B * S;                        // 4096
    const float scale = 1.0f / sqrtf(192.0f);   // DQ^-0.5 (full q head dim)

    f32_to_bf16_kernel<<<(M * HID) / (256 * 8), 256, 0, stream>>>(
        hs, hsb, (long)M * HID);

    // q_nope = hs @ wq[:, head*192 : head*192+128]  -> [4096, 4096]
    gemm_bf16_kernel<bf16><<<dim3(4096 / G_BN, M / G_BM), 256, 0, stream>>>(
        hsb, HID, wq, 6144, /*grp_wstride=*/192, qb, 4096, HID);

    // kv = hs @ wkv[:, :8192]  -> [4096, 8192]  (identity column mapping)
    gemm_bf16_kernel<bf16><<<dim3(8192 / G_BN, M / G_BM), 256, 0, stream>>>(
        hsb, HID, wkv, 8256, /*grp_wstride=*/128, kvb, 8192, HID);

    // attention -> attn [4096, 4096]
    mla_attention_kernel<<<dim3(S / 128, 32, B), 256, 0, stream>>>(
        qb, kvb, attn, scale, S);

    // out = attn @ wo -> [4096, 4096] f32
    gemm_bf16_kernel<float><<<dim3(4096 / G_BN, M / G_BM), 256, 0, stream>>>(
        attn, 4096, wo, 4096, /*grp_wstride=*/128, out, HID, 4096);
}
